// MyConv2d_package_84825604096307
// MI455X (gfx1250) — matrix operand order repaired and checked by reading, not compiled, not run on hardware
//
#include <hip/hip_runtime.h>
#include <hip/hip_bf16.h>

// ---------------------------------------------------------------------------
// Implicit-GEMM 3x3 conv (stride 1, pad 1), NCHW fp32, bf16 hi/lo split WMMA
// (3x v_wmma_f32_16x16x32_bf16 per 16x16x32 slice, fp32 accumulate).
//
// K-order TAP-MAJOR: k' = tap*128 + c. Weights pre-split to bf16 hi/lo
// dword-packed in d_ws (prep kernel). A-tile staging then needs NO transform,
// so it uses the CDNA5 async data mover: global_load_async_to_lds_b128
// (ASYNCcnt-tracked, double-buffered LDS, s_wait_asynccnt pipelining).
// B-tile (im2col) needs an fp32->bf16 split, so it stays on the
// register-pipelined gather path.
// LDS row stride 20 dwords (80B): 16B-aligned uint4, gcd(20,64)=4 ->
// conflict-free ds_load_b128 fragment reads.
// Grid: M-tile index varies FASTEST so the two blocks sharing one B-tile are
// temporally adjacent -> input streamed from HBM once, second pass hits L2.
// ---------------------------------------------------------------------------

typedef __attribute__((ext_vector_type(16))) __bf16 v16bf;
typedef __attribute__((ext_vector_type(8)))  float  v8f;

#define N_IMG   32
#define C_IN    128
#define H_IN    112
#define W_IN    112
#define K_OUT   256
#define K_DIM   1152              // C_IN * 9
#define PIX     (H_IN * W_IN)     // 12544 (divisible by 128)
#define NPIX    (N_IMG * PIX)     // 401408

#define MT 128                    // M tile (out channels)
#define NT 128                    // N tile (pixels)
#define KT 32                     // K step (stays within one tap)
#define RS 20                     // LDS row stride in dwords (80B)
#define KD2 (K_DIM / 2)           // 576 dwords per weight row

union FragBF { v16bf v; uint4 q[2]; };

__device__ __forceinline__ unsigned bf16_rne(unsigned u) {
  return (u + 0x7FFFu + ((u >> 16) & 1u)) >> 16;
}

__device__ __forceinline__ void split_pack2(float f0, float f1,
                                            unsigned &hd, unsigned &ld) {
  unsigned u0 = __float_as_uint(f0);
  unsigned h0 = bf16_rne(u0);
  unsigned l0 = bf16_rne(__float_as_uint(f0 - __uint_as_float(h0 << 16)));
  unsigned u1 = __float_as_uint(f1);
  unsigned h1 = bf16_rne(u1);
  unsigned l1 = bf16_rne(__float_as_uint(f1 - __uint_as_float(h1 << 16)));
  hd = h0 | (h1 << 16);
  ld = l0 | (l1 << 16);
}

// Low 32 bits of a generic pointer to LDS = LDS byte address.
__device__ __forceinline__ unsigned lds_addr32(const void* p) {
  return (unsigned)(unsigned long long)p;
}

// CDNA5 async path: DMA 16B per lane, global -> LDS, ASYNCcnt-tracked.
__device__ __forceinline__ void async_b128(unsigned lds_off, unsigned voff_bytes,
                                           const unsigned* sbase) {
  asm volatile("global_load_async_to_lds_b128 %0, %1, %2"
               :: "v"(lds_off), "v"(voff_bytes), "s"(sbase) : "memory");
}

// ---- prep: weights -> tap-major, bf16 hi/lo split, dword-packed ----
__global__ __launch_bounds__(256)
void prep_weights(const float* __restrict__ Wt,
                  unsigned* __restrict__ wsAh, unsigned* __restrict__ wsAl) {
  const int e = blockIdx.x * 256 + threadIdx.x;   // 0 .. 256*576-1
  const int m = e / KD2;
  const int d = e - m * KD2;
  const int k0  = 2 * d;
  const int tap = k0 >> 7;
  const int c0  = k0 & 127;
  const float f0 = Wt[m * K_DIM + c0 * 9 + tap];
  const float f1 = Wt[m * K_DIM + (c0 + 1) * 9 + tap];
  unsigned hd, ld;
  split_pack2(f0, f1, hd, ld);
  wsAh[e] = hd;
  wsAl[e] = ld;
}

__global__ __launch_bounds__(256)
void conv3x3_igemm_bf16x3(const float* __restrict__ In,
                          const unsigned* __restrict__ wsAh,
                          const unsigned* __restrict__ wsAl,
                          const float* __restrict__ Bias,
                          float* __restrict__ Out) {
  // A double-buffered (async-filled), B single (reg-pipelined). 61.4 KB total.
  __shared__ __align__(16) unsigned sAh[2][MT * RS];
  __shared__ __align__(16) unsigned sAl[2][MT * RS];
  __shared__ __align__(16) unsigned sBh[NT * RS];
  __shared__ __align__(16) unsigned sBl[NT * RS];

  const int t  = threadIdx.x;
  const int m0 = blockIdx.x * MT;      // M-tile: FASTEST-varying block index
  const int p0 = blockIdx.y * NT;      // pixel tile

  // ---- A async map: 128 rows x 4 16B-chunks x {hi,lo}; thread covers 2 chunks
  const int mA0 = t >> 2;              // rows 0..63
  const int mA1 = (t + 256) >> 2;      // rows 64..127
  const int qA  = t & 3;               // 16B chunk within row

  // ---- B staging map ----
  const int pB  = t & 127;             // pixel column (fixed)
  const int kq0 = t >> 7;              // dword col = 2*i + kq0
  const int gpix  = p0 + pB;
  const int nimgB = gpix / PIX;
  const int remB  = gpix - nimgB * PIX;
  const int ohB   = remB / W_IN;
  const int owB   = remB - ohB * W_IN;
  const float* InB = In + (nimgB * C_IN) * PIX;

  // ---- wave tiling ----
  const int lane = t & 31;
  const int wv   = t >> 5;
  const int wm   = wv & 1;             // 2 waves along M (64 rows)
  const int wn   = wv >> 1;            // 4 waves along N (32 cols)
  const int lrow = lane & 15;
  const int lhi  = lane >> 4;

  // Issue the whole A hi/lo tile for step kc into LDS buffer `buf`:
  // 4 wave-level async b128 instructions (ASYNCcnt += 4).
  auto issueA = [&](int kc, int buf) {
    const unsigned* gh = wsAh + m0 * KD2 + (kc >> 1);   // uniform bases
    const unsigned* gl = wsAl + m0 * KD2 + (kc >> 1);
    const unsigned vo0 = (unsigned)(mA0 * KD2 + qA * 4) * 4u;  // bytes
    const unsigned vo1 = (unsigned)(mA1 * KD2 + qA * 4) * 4u;
    async_b128(lds_addr32(&sAh[buf][mA0 * RS + qA * 4]), vo0, gh);
    async_b128(lds_addr32(&sAh[buf][mA1 * RS + qA * 4]), vo1, gh);
    async_b128(lds_addr32(&sAl[buf][mA0 * RS + qA * 4]), vo0, gl);
    async_b128(lds_addr32(&sAl[buf][mA1 * RS + qA * 4]), vo1, gl);
  };

  // ---- register-pipelined B staging ----
  float b0[8], b1[8];
  auto loadB = [&](int kc) {
    const int tap = kc >> 7;           // fixed for the whole step
    const int r   = tap / 3;
    const int s   = tap - 3 * r;
    const int ih  = ohB + r - 1;
    const int iw  = owB + s - 1;
    const bool valid = ((unsigned)ih < (unsigned)H_IN) &&
                       ((unsigned)iw < (unsigned)W_IN);
    const int cbase = kc & 127;
    const float* p = InB + ih * W_IN + iw;
    #pragma unroll
    for (int i = 0; i < 8; ++i) {
      const int c = cbase + (i * 2 + kq0) * 2;
      b0[i] = valid ? p[c * PIX]       : 0.0f;
      b1[i] = valid ? p[(c + 1) * PIX] : 0.0f;
    }
  };
  auto storeB = [&]() {
    #pragma unroll
    for (int i = 0; i < 8; ++i) {
      const int kp = i * 2 + kq0;
      unsigned hd, ld;
      split_pack2(b0[i], b1[i], hd, ld);
      sBh[pB * RS + kp] = hd;
      sBl[pB * RS + kp] = ld;
    }
  };

  v8f acc[4][2] = {};

  int bufA = 0;
  issueA(0, 0);
  loadB(0);

  #pragma unroll 1
  for (int kc = 0; kc < K_DIM; kc += KT) {
    __syncthreads();                   // prior iteration's LDS reads done
    storeB();                          // B regs -> LDS
    if (kc + KT < K_DIM) {
      issueA(kc + KT, bufA ^ 1);       // async DMA next A tile (other buffer)
      loadB(kc + KT);                  // next B into regs
      // 4 newer async ops in flight; in-order completion => current buffer done
      asm volatile("s_wait_asynccnt 0x4" ::: "memory");
    } else {
      asm volatile("s_wait_asynccnt 0x0" ::: "memory");
    }
    __syncthreads();                   // tile visible to all waves

    // ---- fragment loads: conflict-free ds_load_b128 ----
    FragBF ah[4], al[4];
    #pragma unroll
    for (int sm = 0; sm < 4; ++sm) {
      const int row = (wm * 64 + sm * 16 + lrow) * RS + lhi * 4;
      ah[sm].q[0] = *(const uint4*)&sAh[bufA][row];
      ah[sm].q[1] = *(const uint4*)&sAh[bufA][row + 8];
      al[sm].q[0] = *(const uint4*)&sAl[bufA][row];
      al[sm].q[1] = *(const uint4*)&sAl[bufA][row + 8];
    }
    FragBF bh[2], bl[2];
    #pragma unroll
    for (int sn = 0; sn < 2; ++sn) {
      const int col = (wn * 32 + sn * 16 + lrow) * RS + lhi * 4;
      bh[sn].q[0] = *(const uint4*)&sBh[col];
      bh[sn].q[1] = *(const uint4*)&sBh[col + 8];
      bl[sn].q[0] = *(const uint4*)&sBl[col];
      bl[sn].q[1] = *(const uint4*)&sBl[col + 8];
    }

    // ---- 3-term split MMA: hi*hi + lo*hi + hi*lo ----
    #pragma unroll
    for (int sm = 0; sm < 4; ++sm) {
      #pragma unroll
      for (int sn = 0; sn < 2; ++sn) {
        acc[sm][sn] = __builtin_amdgcn_wmma_f32_16x16x32_bf16(
            false, ah[sm].v, false, bh[sn].v, (short)0, acc[sm][sn], false, false);
        acc[sm][sn] = __builtin_amdgcn_wmma_f32_16x16x32_bf16(
            false, al[sm].v, false, bh[sn].v, (short)0, acc[sm][sn], false, false);
        acc[sm][sn] = __builtin_amdgcn_wmma_f32_16x16x32_bf16(
            false, ah[sm].v, false, bl[sn].v, (short)0, acc[sm][sn], false, false);
      }
    }
    bufA ^= 1;
  }

  // ---- epilogue: D 16x16 f32: lane -> col n=lane&15, VGPR v -> m=v+8*(lane>>4)
  #pragma unroll
  for (int sm = 0; sm < 4; ++sm) {
    #pragma unroll
    for (int sn = 0; sn < 2; ++sn) {
      const int pg   = p0 + wn * 32 + sn * 16 + lrow;
      const int nimg = pg / PIX;
      const int rem2 = pg - nimg * PIX;
      #pragma unroll
      for (int vv = 0; vv < 8; ++vv) {
        const int m = m0 + wm * 64 + sm * 16 + lhi * 8 + vv;
        Out[(nimg * K_OUT + m) * PIX + rem2] = acc[sm][sn][vv] + Bias[m];
      }
    }
  }
}

extern "C" void kernel_launch(void* const* d_in, const int* in_sizes, int n_in,
                              void* d_out, int out_size, void* d_ws, size_t ws_size,
                              hipStream_t stream) {
  (void)in_sizes; (void)n_in; (void)out_size; (void)ws_size;
  const float* in   = (const float*)d_in[0];   // 32*128*112*112
  const float* wt   = (const float*)d_in[1];   // 256*128*3*3
  const float* bias = (const float*)d_in[2];   // 256
  float* out = (float*)d_out;                  // 32*256*112*112

  unsigned* wsAh = (unsigned*)d_ws;            // 256*576 dwords
  unsigned* wsAl = wsAh + K_OUT * KD2;         // total ~1.2 MB of scratch

  prep_weights<<<dim3(K_OUT * KD2 / 256), dim3(256), 0, stream>>>(wt, wsAh, wsAl);

  // M-tile fastest => blocks sharing a B-tile are adjacent => input hits L2.
  dim3 grid(K_OUT / MT, NPIX / NT, 1);         // (2, 3136)
  dim3 block(256, 1, 1);                       // 8 wave32
  conv3x3_igemm_bf16x3<<<grid, block, 0, stream>>>(in, wsAh, wsAl, bias, out);
}
